// MultiHeadAttention_36043365548445
// MI455X (gfx1250) — matrix operand order repaired and checked by reading, not compiled, not run on hardware
//
#include <hip/hip_runtime.h>
#include <hip/hip_bf16.h>

// ---------------------------------------------------------------------------
// MHA forward for MI455X (gfx1250): bf16 WMMA, vectorized LDS fragments,
// TDM (tensor_load_to_lds) staging of K tiles, fused attention.
// B=4, S=1024, D=1024, H=16, dh=64.
// ---------------------------------------------------------------------------

typedef __attribute__((ext_vector_type(16))) __bf16 v16bf;
typedef __attribute__((ext_vector_type(8)))  __bf16 v8bf;
typedef __attribute__((ext_vector_type(8)))  float  v8f;

typedef unsigned int uint4v __attribute__((ext_vector_type(4)));
typedef int          int8v  __attribute__((ext_vector_type(8)));
typedef int          int4v  __attribute__((ext_vector_type(4)));

#define DIM_B  4
#define DIM_S  1024
#define DIM_D  1024
#define DIM_H  16
#define DIM_DH 64
#define TQ     32      // q rows per workgroup in attention kernel
#define TK     128     // k/v rows staged per iteration

// LDS layout of the attention kernel (byte offsets into dynamic LDS)
#define SMEM_SS_OFF   0                                   // [TQ][S] f32   128 KB
#define SMEM_SP_OFF   (TQ * DIM_S * 4)                    // [TQ][S] bf16   64 KB
#define SMEM_SQ_OFF   (TQ * DIM_S * 6)                    // [TQ][64] bf16   4 KB
#define SMEM_SKV_OFF  (SMEM_SQ_OFF + TQ * DIM_DH * 2)     // [TK][64] bf16  16 KB
#define SMEM_RED_OFF  (SMEM_SKV_OFF + TK * DIM_DH * 2)    // [TQ][8] f32     1 KB
#define SMEM_TOTAL    (SMEM_RED_OFF + TQ * 8 * 4)

static __device__ __forceinline__ __bf16 to_bf16(float f) {
    unsigned u = __float_as_uint(f);
    unsigned r = (u + 0x7FFFu + ((u >> 16) & 1u)) >> 16;   // round-nearest-even
    unsigned short s = (unsigned short)r;
    return __builtin_bit_cast(__bf16, s);
}

static __device__ __forceinline__ v8f wmma_bf16(v16bf a, v16bf b, v8f c) {
    return __builtin_amdgcn_wmma_f32_16x16x32_bf16(false, a, false, b,
                                                   (short)0, c, false, false);
}

// A fragment (16x32 bf16), tile row-major [m][k], stride ld (elements).
// Lane layout: m = lane&15 ; K runs kadd..kadd+15
// with kadd = 16*(lane>=16)  -> two contiguous 16B LDS loads.
static __device__ __forceinline__ v16bf load_a_frag(const __bf16* base, int ld, int lane) {
    int m    = lane & 15;
    int kadd = (lane >> 4) << 4;
    const __bf16* p = base + m * ld + kadd;
    v8bf lo = *(const v8bf*)(p);
    v8bf hi = *(const v8bf*)(p + 8);
    return __builtin_shufflevector(lo, hi, 0, 1, 2, 3, 4, 5, 6, 7,
                                           8, 9, 10, 11, 12, 13, 14, 15);
}

// B fragment (32x16 bf16) from a tile stored [n][k] with stride ld (elements).
// Lane layout: n = lane&15 ; K runs koff..koff+15 with koff = 16*(lane>=16)
// -> one contiguous 32B run per lane (two 16B LDS loads).
static __device__ __forceinline__ v16bf load_b_frag(const __bf16* baseN, int ld, int lane) {
    int n    = lane & 15;
    int koff = (lane >> 4) << 4;
    const __bf16* p = baseN + n * ld + koff;
    v8bf lo = *(const v8bf*)(p);
    v8bf hi = *(const v8bf*)(p + 8);
    return __builtin_shufflevector(lo, hi, 0, 1, 2, 3, 4, 5, 6, 7,
                                           8, 9, 10, 11, 12, 13, 14, 15);
}

// ---------------------------------------------------------------------------
// Kernel 1: QKV projections.  Out_z = X_z @ W_z + b_z, stored as bf16 in ws.
// M = B*S = 4096, N = K = 1024.  grid = (N/128, M/128, 3), block = 256 (8 waves).
// A tile staged row-major [m][k]; W tile staged TRANSPOSED [n][k] so B
// fragments are contiguous per lane.
// ---------------------------------------------------------------------------
__global__ __launch_bounds__(256)
void mha_qkv_proj_kernel(const float* __restrict__ Xq, const float* __restrict__ Xk,
                         const float* __restrict__ Xv,
                         const float* __restrict__ Wq, const float* __restrict__ Wk,
                         const float* __restrict__ Wv,
                         const float* __restrict__ bq, const float* __restrict__ bk,
                         const float* __restrict__ bv,
                         __bf16* __restrict__ OutBase) {
    const int M = DIM_B * DIM_S;   // 4096
    const int N = DIM_D;           // 1024
    const int K = DIM_D;           // 1024

    int z = blockIdx.z;
    const float* X    = (z == 0) ? Xq : (z == 1) ? Xk : Xv;
    const float* W    = (z == 0) ? Wq : (z == 1) ? Wk : Wv;
    const float* bias = (z == 0) ? bq : (z == 1) ? bk : bv;
    __bf16* Out = OutBase + (size_t)z * M * N;

    int n0 = blockIdx.x * 128;
    int m0 = blockIdx.y * 128;
    int t    = threadIdx.x;
    int lane = t & 31;
    int wv   = t >> 5;          // 0..7
    int wr   = wv >> 1;         // 0..3  -> rows [wr*32, wr*32+32)
    int wc   = wv & 1;          // 0..1  -> cols [wc*64, wc*64+64)

    __shared__ __align__(16) __bf16 sA [128 * 32];   // [m][k]
    __shared__ __align__(16) __bf16 sBt[128 * 32];   // [n][k]  (transposed W tile)

    v8f acc[2][4] = {};

    for (int kk = 0; kk < K; kk += 32) {
        // stage A: 128x32, 16 contiguous floats per thread, row-major
        {
            int row = t >> 1, cb = (t & 1) * 16;
            const float* src = X + (size_t)(m0 + row) * K + kk + cb;
            __bf16* dst = sA + row * 32 + cb;
#pragma unroll
            for (int i = 0; i < 16; ++i) dst[i] = to_bf16(src[i]);
            if (kk + 32 < K) __builtin_prefetch(src + 32, 0, 0);
        }
        // stage W transposed: read 16 contiguous floats of row k, scatter to [n][k]
        {
            int row = t >> 3, cb = (t & 7) * 16;       // row = k idx, cb = n offset
            const float* src = W + (size_t)(kk + row) * N + n0 + cb;
#pragma unroll
            for (int i = 0; i < 16; ++i) sBt[(cb + i) * 32 + row] = to_bf16(src[i]);
            if (kk + 32 < K) __builtin_prefetch(src + (size_t)32 * N, 0, 0);
        }
        __syncthreads();

        v16bf afrag[2];
#pragma unroll
        for (int i = 0; i < 2; ++i)
            afrag[i] = load_a_frag(sA + (wr * 32 + i * 16) * 32, 32, lane);
#pragma unroll
        for (int j = 0; j < 4; ++j) {
            v16bf bfrag = load_b_frag(sBt + (wc * 64 + j * 16) * 32, 32, lane);
#pragma unroll
            for (int i = 0; i < 2; ++i)
                acc[i][j] = wmma_bf16(afrag[i], bfrag, acc[i][j]);
        }
        __syncthreads();
    }

    // epilogue: + bias, convert to bf16, store
    int nlo = lane & 15;
    int mhi = (lane >> 4) * 8;
#pragma unroll
    for (int i = 0; i < 2; ++i) {
#pragma unroll
        for (int j = 0; j < 4; ++j) {
            int ncol = n0 + wc * 64 + j * 16 + nlo;
            float bval = bias[ncol];
#pragma unroll
            for (int r = 0; r < 8; ++r) {
                int mrow = m0 + wr * 32 + i * 16 + r + mhi;
                Out[(size_t)mrow * N + ncol] = to_bf16(acc[i][j][r] + bval);
            }
        }
    }
}

// ---------------------------------------------------------------------------
// TDM: issue a 2D tile load (TK rows x DIM_DH bf16, row stride DIM_D) from
// global into LDS at byte offset lds_off.  Wave-level op (EXEC ignored);
// call from a wave-uniform branch.  D# per CDNA5 ISA:
//   group0: count=1 | lds_addr | global_addr[56:0] | type=2
//   group1: data_size=2B, tensor_dim0/1, tile_dim0/1, dim0_stride
// This toolchain exposes the 6-arg builtin:
//   (uint32x4 g0, int32x8 g1, int32x4 g2, int32x4 g3, int32x8 g4, i32 cpol)
// ---------------------------------------------------------------------------
static __device__ __forceinline__ void tdm_load_tile_bf16(const __bf16* gptr,
                                                          unsigned lds_off) {
    unsigned long long ga = (unsigned long long)(const void*)gptr;
    uint4v g0;
    g0[0] = 1u;                                            // count=1, user mode
    g0[1] = lds_off;                                       // lds_addr (bytes)
    g0[2] = (unsigned)(ga & 0xFFFFFFFFu);                  // global_addr[31:0]
    g0[3] = (unsigned)((ga >> 32) & 0x1FFFFFFu) | (2u << 30); // [56:32] | type=2

    int8v g1;
    g1[0] = (int)(1u << 16);                               // data_size=1 (2 bytes)
    g1[1] = (int)((unsigned)DIM_D << 16);                  // tensor_dim0[15:0] @ [31:16]
    g1[2] = (int)((unsigned)DIM_S << 16);                  // dim0[31:16]=0 | tensor_dim1[15:0]
    g1[3] = (int)((unsigned)DIM_DH << 16);                 // dim1[31:16]=0 | tile_dim0=64
    g1[4] = TK;                                            // tile_dim1=128, tile_dim2=0
    g1[5] = DIM_D;                                         // tensor_dim0_stride[31:0]
    g1[6] = 0;                                             // stride[47:32] | dim1_stride lo
    g1[7] = 0;

    int4v gz4 = {0, 0, 0, 0};
    int8v gz8 = {0, 0, 0, 0, 0, 0, 0, 0};
    __builtin_amdgcn_tensor_load_to_lds(g0, g1, gz4, gz4, gz8, 0);
}

// ---------------------------------------------------------------------------
// Kernel 2: fused attention per (b, h, q-tile of 32 rows).
//   scores = (Q K^T) * 1/sqrt(dh) -> 32x1024 f32 strip in LDS
//   softmax in LDS, attn written once to global, bf16 copy kept in LDS
//   context = attn @ V via bf16 WMMA.
// grid = (S/TQ, H, B), block = 256 (8 waves). Dynamic LDS ~213 KB (<320 KB/WGP).
// K tiles staged by the Tensor Data Mover; V tiles staged transposed by VALU.
// ---------------------------------------------------------------------------
__global__ __launch_bounds__(256)
void mha_attn_fused_kernel(const __bf16* __restrict__ Qb, const __bf16* __restrict__ Kb,
                           const __bf16* __restrict__ Vb,
                           float* __restrict__ ctx, float* __restrict__ attnOut) {
    const int S = DIM_S, D = DIM_D;

    int q0 = blockIdx.x * TQ;
    int h  = blockIdx.y;
    int b  = blockIdx.z;
    int t    = threadIdx.x;
    int lane = t & 31;
    int wv   = t >> 5;

    extern __shared__ __align__(16) char smem[];
    float*  sS   = (float*)(smem + SMEM_SS_OFF);    // [TQ][S]
    __bf16* sP   = (__bf16*)(smem + SMEM_SP_OFF);   // [TQ][S]
    __bf16* sQ   = (__bf16*)(smem + SMEM_SQ_OFF);   // [TQ][64]
    __bf16* sKV  = (__bf16*)(smem + SMEM_SKV_OFF);  // [TK][64] (K) / [64][TK] (V^T)
    float*  sRed = (float*)(smem + SMEM_RED_OFF);   // [TQ][8]

    // ---- stage Q tile (32 x 64 bf16), 8 contiguous elems per thread
    {
        int idx = t * 8;
        int row = idx >> 6, col = idx & 63;
        const __bf16* src = Qb + ((size_t)(b * S + q0 + row)) * D + h * DIM_DH + col;
#pragma unroll
        for (int i = 0; i < 8; ++i) sQ[row * DIM_DH + col + i] = src[i];
    }
    __syncthreads();

    // ---- Phase B: scores = Q K^T * 0.125 into sS.  K tile via TDM.
    for (int k0 = 0; k0 < S; k0 += TK) {
        if (wv == 0) {   // wave-uniform: one TDM op per workgroup per tile
            tdm_load_tile_bf16(Kb + ((size_t)(b * S + k0)) * D + h * DIM_DH,
                               SMEM_SKV_OFF);
            __builtin_amdgcn_s_wait_tensorcnt(0);
        }
        __syncthreads();

        // 32x128 score subtile = 16 tiles of 16x16; 2 per wave.
        // K tile [kv_row][dh] is already the [n][k] layout B=K^T wants.
#pragma unroll
        for (int u = 0; u < 2; ++u) {
            int ti = wv * 2 + u;
            int mi = ti >> 3, ni = ti & 7;
            v8f acc = {};
#pragma unroll
            for (int ks = 0; ks < DIM_DH; ks += 32) {
                v16bf a  = load_a_frag(sQ + (mi * 16) * DIM_DH + ks, DIM_DH, lane);
                v16bf bb = load_b_frag(sKV + (ni * 16) * DIM_DH + ks, DIM_DH, lane);
                acc = wmma_bf16(a, bb, acc);
            }
            int nlo = lane & 15, mhi = (lane >> 4) * 8;
#pragma unroll
            for (int r = 0; r < 8; ++r)
                sS[(mi * 16 + r + mhi) * S + (k0 + ni * 16 + nlo)] = acc[r] * 0.125f;
        }
        __syncthreads();
    }

    // ---- Phase C: softmax (8 threads per row), write attn, keep bf16 in sP
    {
        int row = t >> 3, sub = t & 7;
        float* rowp = sS + row * S;

        float lmax = -3.4e38f;
#pragma unroll 8
        for (int jj = 0; jj < 128; ++jj) lmax = fmaxf(lmax, rowp[sub + jj * 8]);
        sRed[row * 8 + sub] = lmax;
        __syncthreads();
        float rmax = sRed[row * 8];
#pragma unroll
        for (int i = 1; i < 8; ++i) rmax = fmaxf(rmax, sRed[row * 8 + i]);
        __syncthreads();

        float lsum = 0.f;
#pragma unroll 8
        for (int jj = 0; jj < 128; ++jj) {
            int j = sub + jj * 8;
            float e = __expf(rowp[j] - rmax);
            rowp[j] = e;
            lsum += e;
        }
        sRed[row * 8 + sub] = lsum;
        __syncthreads();
        float rsum = 0.f;
#pragma unroll
        for (int i = 0; i < 8; ++i) rsum += sRed[row * 8 + i];
        float inv = 1.0f / rsum;

        // attn layout: (h*B + b, q, k)
        float* aout = attnOut + (((size_t)(h * DIM_B + b)) * S + (q0 + row)) * S;
        __bf16* prow = sP + row * S;
#pragma unroll 8
        for (int jj = 0; jj < 128; ++jj) {
            int j = sub + jj * 8;
            float p = rowp[j] * inv;
            aout[j] = p;
            prow[j] = to_bf16(p);
        }
    }
    __syncthreads();

    // ---- Phase D: context = attn @ V ; 8 tiles of 16x16 (2 M x 4 N), 1/wave.
    // V tile staged TRANSPOSED: sVt[n(0..63)][k(0..127)] so B frags are contiguous.
    __bf16* sVt = sKV;
    int mi = wv >> 2, ni = wv & 3;
    v8f acc = {};
    for (int k0 = 0; k0 < S; k0 += TK) {
        {
            int row = t >> 1, cb = (t & 1) * 32;   // row = k idx, cb = n offset
            const __bf16* src = Vb + ((size_t)(b * S + k0 + row)) * D + h * DIM_DH + cb;
#pragma unroll
            for (int i = 0; i < 32; ++i) sVt[(cb + i) * TK + row] = src[i];
        }
        __syncthreads();
#pragma unroll
        for (int ks = 0; ks < TK; ks += 32) {
            v16bf a  = load_a_frag(sP + (mi * 16) * S + k0 + ks, S, lane);
            v16bf bb = load_b_frag(sVt + (ni * 16) * TK + ks, TK, lane);
            acc = wmma_bf16(a, bb, acc);
        }
        __syncthreads();
    }

    int nlo = lane & 15, mhi = (lane >> 4) * 8;
#pragma unroll
    for (int r = 0; r < 8; ++r) {
        int m   = q0 + mi * 16 + r + mhi;
        int col = h * DIM_DH + ni * 16 + nlo;
        ctx[((size_t)b * S + m) * D + col] = acc[r];
    }
}

// ---------------------------------------------------------------------------
extern "C" void kernel_launch(void* const* d_in, const int* in_sizes, int n_in,
                              void* d_out, int out_size, void* d_ws, size_t ws_size,
                              hipStream_t stream) {
    const float* q  = (const float*)d_in[0];
    const float* k  = (const float*)d_in[1];
    const float* v  = (const float*)d_in[2];
    const float* Wq = (const float*)d_in[3];
    const float* bq = (const float*)d_in[4];
    const float* Wk = (const float*)d_in[5];
    const float* bk = (const float*)d_in[6];
    const float* Wv = (const float*)d_in[7];
    const float* bv = (const float*)d_in[8];

    float* out  = (float*)d_out;
    float* ctx  = out;                                             // (B, S, D)
    float* attn = out + (size_t)DIM_B * DIM_S * DIM_D;             // (H*B, S, S)

    __bf16* ws = (__bf16*)d_ws;
    const size_t mn = (size_t)DIM_B * DIM_S * DIM_D;               // 4M elems
    __bf16* Qb = ws;            // bf16 projected Q
    __bf16* Kb = ws + mn;       // bf16 projected K
    __bf16* Vb = ws + 2 * mn;   // bf16 projected V

    // Kernel 1: QKV projections (N/128, M/128, 3)
    dim3 g1(DIM_D / 128, (DIM_B * DIM_S) / 128, 3);
    mha_qkv_proj_kernel<<<g1, dim3(256), 0, stream>>>(q, k, v, Wq, Wk, Wv,
                                                      bq, bk, bv, ws);

    // Kernel 2: fused attention (S/TQ, H, B)
    dim3 g2(DIM_S / TQ, DIM_H, DIM_B);
    mha_attn_fused_kernel<<<g2, dim3(256), (size_t)SMEM_TOTAL, stream>>>(
        Qb, Kb, Vb, ctx, attn);
}
